// TAGnn_50732153700640
// MI455X (gfx1250) — matrix operand order repaired and checked by reading, not compiled, not run on hardware
//
#include <hip/hip_runtime.h>
#include <hip/hip_bf16.h>

// ---------------- problem constants ----------------
#define B_    32
#define N_    716
#define T_    12
#define TM1   11
#define P_    12
#define HID   64
#define DD    10
#define OUTK  128
#define HOUR_ 288
#define DAY_  7
#define NN    (716*716)      /* 512656 */
#define NTILE 45             /* ceil(716/16) */
#define NJ    23             /* ceil(716/32) */
#define TH    (T_*HID)       /* 768 */

typedef __attribute__((ext_vector_type(16))) _Float16 v16h;
typedef __attribute__((ext_vector_type(8)))  float    v8f;

// ---------------- WMMA fragment layout helpers ----------------
// A matrix 16(M)x32(K): lane l: m=l&15, hi=l>>4 ; element e -> k=(e&7)+8*hi+16*(e>>3)
// inverse: element (m,k) -> lane=(m&15)+16*((k>>3)&1), e=(k&7)+8*((k>>4)&1)
__device__ __forceinline__ int afrag_idx(int m, int k) {   // k in [0,32)
  return ((m & 15) + 16 * ((k >> 3) & 1)) * 16 + (k & 7) + 8 * ((k >> 4) & 1);
}
// B matrix 32(K)x16(N): lane l: n=l&15, hi=l>>4 ; element e -> k=(e&7)+8*hi+16*(e>>3)
// inverse: element (k,n) -> lane=(n&15)+16*((k>>3)&1), e=(k&7)+8*((k>>4)&1)
__device__ __forceinline__ int bfrag_idx(int k, int n) {   // k in [0,32)
  return ((n & 15) + 16 * ((k >> 3) & 1)) * 16 + (k & 7) + 8 * ((k >> 4) & 1);
}
// lane's fragment = 16 contiguous halves (32B) -> 2x b128 loads
__device__ __forceinline__ v16h frag_ld(const _Float16* tile) {
  return *(const v16h*)(tile + (threadIdx.x & 31) * 16);
}
__device__ __forceinline__ v8f wmma16(const v16h a, const v16h b, const v8f c) {
  return __builtin_amdgcn_wmma_f32_16x16x32_f16(false, a, false, b, (short)0, c,
                                                false, false);
}
// C/D layout: lane l: n=l&15 ; vgpr r -> m = r + 8*(l>>4)

// ---------------- weight conversion into B-fragment order ----------------
// src: mats x K x C row-major f32 -> dst: mats x (K/32) x (C/16) x 32 x 16 f16
__global__ void cvt_bfrag_kernel(const float* __restrict__ src,
                                 _Float16* __restrict__ dst,
                                 int K, int C, int total) {
  int idx = blockIdx.x * 256 + threadIdx.x;
  if (idx >= total) return;
  int kc = K * C;
  int mat = idx / kc;
  int rem = idx - mat * kc;
  int k = rem / C, c = rem - (rem / C) * C;
  int tile = (k >> 5) * (C >> 4) + (c >> 4);
  int dsz = (K >> 5) * (C >> 4) * 512;
  dst[(size_t)mat * dsz + tile * 512 + bfrag_idx(k & 31, c & 15)] = (_Float16)src[idx];
}

// W_tcn (O=64, I=64, 1, TAP=3) -> B-fragment order with K = tau*64+i, C = o
__global__ void cvt_wtcn_frag_kernel(const float* __restrict__ w,
                                     _Float16* __restrict__ dst) {
  int idx = blockIdx.x * 256 + threadIdx.x;      // 64*64*3
  if (idx < 64 * 64 * 3) {
    int o = idx / 192;
    int rem = idx % 192;
    int i = rem / 3, tau = rem % 3;
    int k = tau * HID + i, c = o;
    int tile = (k >> 5) * 4 + (c >> 4);
    dst[tile * 512 + bfrag_idx(k & 31, c & 15)] = (_Float16)w[idx];
  }
}

// ---------------- hour/day embedding: hd[b][t][h] ----------------
__global__ void hd_kernel(const float* __restrict__ t_hour,
                          const float* __restrict__ t_day,
                          const float* __restrict__ Wh, const float* __restrict__ bh,
                          const float* __restrict__ Wd, const float* __restrict__ bd,
                          float* __restrict__ hd) {
  int bt = blockIdx.x;            // 0..383
  int h = threadIdx.x;            // 0..63
  const float* th = t_hour + (size_t)bt * HOUR_;
  const float* td = t_day + (size_t)bt * DAY_;
  float sh = bh[h];
  for (int j = 0; j < HOUR_; ++j) sh += th[j] * Wh[j * HID + h];
  float sd = bd[h];
  for (int j = 0; j < DAY_; ++j) sd += td[j] * Wd[j * HID + h];
  hd[(size_t)bt * HID + h] = fmaxf(sh, 0.f) + fmaxf(sd, 0.f);
}

// ---------------- adjacency-MLP hidden: h1[b][t][d], h1l[b][d] ----------------
__global__ void __launch_bounds__(128) h1_kernel(
    const float* __restrict__ x, const float* __restrict__ aml_W1,
    const float* __restrict__ aml_b1, const float* __restrict__ last_W1,
    const float* __restrict__ last_b1, float* __restrict__ h1,
    float* __restrict__ h1l) {
  int b = blockIdx.x, t = blockIdx.y;   // t==11 -> last path
  int tid = threadIdx.x;
  __shared__ float red[128][DD];
  float acc[DD];
#pragma unroll
  for (int d = 0; d < DD; ++d) acc[d] = 0.f;
  const float* xb = x + (size_t)b * N_ * T_;
  if (t < TM1) {
    const float* W = aml_W1 + (size_t)t * (N_ * 2) * DD;
    for (int n = tid; n < N_; n += 128) {
      float x0 = xb[n * T_ + t];
      float x1 = xb[n * T_ + (T_ - 1)];
      const float* w0 = W + (size_t)(2 * n) * DD;
#pragma unroll
      for (int d = 0; d < DD; ++d) acc[d] += x0 * w0[d] + x1 * w0[DD + d];
    }
  } else {
    for (int n = tid; n < N_; n += 128) {
      float xl = xb[n * T_ + (T_ - 1)];
      const float* w = last_W1 + (size_t)n * DD;
#pragma unroll
      for (int d = 0; d < DD; ++d) acc[d] += xl * w[d];
    }
  }
#pragma unroll
  for (int d = 0; d < DD; ++d) red[tid][d] = acc[d];
  __syncthreads();
  for (int s = 64; s > 0; s >>= 1) {
    if (tid < s)
#pragma unroll
      for (int d = 0; d < DD; ++d) red[tid][d] += red[tid + s][d];
    __syncthreads();
  }
  if (tid < DD) {
    float v = red[0][tid];
    if (t < TM1) h1[((size_t)b * TM1 + t) * DD + tid] = v + aml_b1[t * DD + tid];
    else         h1l[(size_t)b * DD + tid] = v + last_b1[tid];
  }
}

// ------- embed: data = relu(acx@W_data+b)@W_data1 + b1 + hd -> embF (A-frag) --
__global__ void __launch_bounds__(128) embed_kernel(
    const float* __restrict__ x, const _Float16* __restrict__ wd1F,
    const float* __restrict__ W_data, const float* __restrict__ b_data,
    const float* __restrict__ b_data1, const float* __restrict__ hd,
    _Float16* __restrict__ embF) {
  int nt = blockIdx.x, bt = blockIdx.y;
  int b = bt / T_, t = bt % T_;
  int n0 = nt * 16;
  int tid = threadIdx.x, lane = tid & 31, w = tid >> 5;
  __shared__ alignas(32) _Float16 sA[2][32][16];   // layer-1 output, A-frag order
  const float* xb = x + (size_t)b * N_ * T_;
  for (int i = tid; i < 16 * 64; i += 128) {
    int m = i >> 6, c = i & 63;
    int n = n0 + m; if (n >= N_) n = N_ - 1;
    float x0 = xb[n * T_ + t];
    float x1 = xb[n * T_ + (T_ - 1)];
    float v = fmaxf(x0 * W_data[c] + x1 * W_data[64 + c] + b_data[c], 0.f);
    (&sA[c >> 5][0][0])[afrag_idx(m, c & 31)] = (_Float16)v;
  }
  __syncthreads();
  v8f acc = {};
#pragma unroll
  for (int kc = 0; kc < 2; ++kc) {
    v16h a = frag_ld(&sA[kc][0][0]);
    v16h bb = frag_ld(wd1F + (size_t)(kc * 4 + w) * 512);
    acc = wmma16(a, bb, acc);
  }
  int hi = lane >> 4, nc = lane & 15;
  int col = w * 16 + nc;
  float badd = b_data1[col];
  float hdv = hd[(size_t)bt * HID + col];
  _Float16* dtile = embF + ((size_t)(bt * NTILE + nt) * 2 + (col >> 5)) * 512;
#pragma unroll
  for (int r = 0; r < 8; ++r) {
    int m = r + 8 * hi;   // pad rows written too (duplicates) -> no poison in embF
    dtile[afrag_idx(m, col & 31)] = (_Float16)(acc[r] + badd + hdv);
  }
}

// ------- TCN: K=192 matmul, A/B straight from fragment-ordered global --------
__global__ void __launch_bounds__(128) tcn_kernel(
    const _Float16* __restrict__ embF, const _Float16* __restrict__ wt2F,
    const float* __restrict__ b_tcn, float* __restrict__ dT32,
    _Float16* __restrict__ dT16F) {
  int nt = blockIdx.x, bt = blockIdx.y;
  int b = bt / T_, t = bt % T_;
  int n0 = nt * 16;
  int tid = threadIdx.x, lane = tid & 31, w = tid >> 5;
  v8f acc = {};
#pragma unroll
  for (int kc = 0; kc < 6; ++kc) {
    int tau = kc >> 1;
    int tt = t + tau - 1;
    if (tt < 0) tt = 0; if (tt > T_ - 1) tt = T_ - 1;   // edge padding
    const _Float16* aT =
        embF + ((size_t)((b * T_ + tt) * NTILE + nt) * 2 + (kc & 1)) * 512;
    v16h a = frag_ld(aT);
    v16h bb = frag_ld(wt2F + (size_t)(kc * 4 + w) * 512);
    acc = wmma16(a, bb, acc);
  }
  int hi = lane >> 4, nc = lane & 15;
  int col = w * 16 + nc;
  float bo = b_tcn[col];
  // dT16F: per (b,t): [NJ][4 ct][32][16] B-fragment tiles (k=j index, n=h)
  _Float16* dtile = dT16F + ((size_t)(bt * NJ + (n0 >> 5)) * 4 + w) * 512;
#pragma unroll
  for (int r = 0; r < 8; ++r) {
    int m = r + 8 * hi, n = n0 + m;
    if (n < N_) {
      float v = acc[r] + bo;
      dT32[((size_t)bt * N_ + n) * HID + col] = v;
      dtile[bfrag_idx(n & 31, col & 15)] = (_Float16)v;   // pads pre-zeroed by memset
    }
  }
}

// ------- fused: adj(tanh rank-10) -> adj@feats -> @gc_K -> GLU gate + res ----
// grid (itile, b, t) with t==11 selecting the "last" path; b fastest for L2 reuse.
__global__ void __launch_bounds__(128) gate_kernel(
    const float* __restrict__ aml_W2, const float* __restrict__ aml_b2,
    const float* __restrict__ last_W2, const float* __restrict__ last_b2,
    const float* __restrict__ h1, const float* __restrict__ h1l,
    const _Float16* __restrict__ gckF, const float* __restrict__ gc_b,
    const _Float16* __restrict__ gcklF, const float* __restrict__ gc_b_last,
    const float* __restrict__ dT32, const _Float16* __restrict__ dT16F,
    _Float16* __restrict__ dfeatF) {
  int it = blockIdx.x, b = blockIdx.y, t = blockIdx.z;
  bool lastp = (t == TM1);
  int i0 = it * 16;
  int tid = threadIdx.x, lane = tid & 31, w = tid >> 5;
  const float* W2 = lastp ? last_W2 : aml_W2 + (size_t)t * DD * NN;
  const float* b2 = lastp ? last_b2 : aml_b2 + (size_t)t * NN;
  float hh[DD];
  {
    const float* hs = lastp ? (h1l + (size_t)b * DD)
                            : (h1 + ((size_t)b * TM1 + t) * DD);
#pragma unroll
    for (int d = 0; d < DD; ++d) hh[d] = hs[d];
  }
  const _Float16* fB = dT16F + (size_t)(b * T_ + t) * NJ * 4 * 512;
  __shared__ alignas(32) _Float16 sAdj[32][16];       // A-frag order
  __shared__ alignas(32) _Float16 sSup[2][32][16];    // A-frag order
  v8f acc = {};
  for (int jc = 0; jc < NJ; ++jc) {
    int j0 = jc * 32;
    __syncthreads();
    for (int idx = tid; idx < 512; idx += 128) {
      int m = idx >> 5, jj = idx & 31;
      int i = i0 + m; if (i >= N_) i = N_ - 1;
      int j = j0 + jj;
      float v = 0.f;
      if (j < N_) {
        size_t base = (size_t)i * N_ + j;
        v = b2[base];
#pragma unroll
        for (int d = 0; d < DD; ++d) v += hh[d] * W2[(size_t)d * NN + base];
        v = tanhf(v);
      }
      (&sAdj[0][0])[afrag_idx(m, jj)] = (_Float16)v;
    }
    __syncthreads();
    v16h a = frag_ld(&sAdj[0][0]);
    v16h bb = frag_ld(fB + (size_t)(jc * 4 + w) * 512);
    acc = wmma16(a, bb, acc);
  }
  // sup tile (16x64) -> LDS in A-fragment order
  __syncthreads();
  {
    int hi = lane >> 4, nc = lane & 15;
    int col = w * 16 + nc;
#pragma unroll
    for (int r = 0; r < 8; ++r)
      (&sSup[col >> 5][0][0])[afrag_idx(r + 8 * hi, col & 31)] = (_Float16)acc[r];
  }
  __syncthreads();
  // o = sup @ gc_K (64x128); wave w owns col tiles w (value) and w+4 (logit)
  const _Float16* Km = lastp ? gcklF : gckF + (size_t)t * 16 * 512;
  const float* gb = lastp ? gc_b_last : gc_b + (size_t)t * OUTK;
  v8f alo = {}, ahi = {};
#pragma unroll
  for (int kc = 0; kc < 2; ++kc) {
    v16h a = frag_ld(&sSup[kc][0][0]);
    v16h blo = frag_ld(Km + (size_t)(kc * 8 + w) * 512);
    alo = wmma16(a, blo, alo);
    v16h bhi = frag_ld(Km + (size_t)(kc * 8 + w + 4) * 512);
    ahi = wmma16(a, bhi, ahi);
  }
  const float* feats32 = dT32 + ((size_t)b * T_ + t) * N_ * HID;
  int hi = lane >> 4, nc = lane & 15;
  int col = w * 16 + nc;
  int f = t * HID + col;
  float bl = gb[col], bh = gb[64 + col];
  _Float16* dtile = dfeatF + ((size_t)(b * NTILE + it) * 24 + (f >> 5)) * 512;
#pragma unroll
  for (int r = 0; r < 8; ++r) {
    int m = r + 8 * hi;
    int i = i0 + m;
    if (i < N_) {
      float o1 = alo[r] + bl;
      float o2 = ahi[r] + bh;
      float fv = feats32[(size_t)i * HID + col];
      float g = o1 * (1.f / (1.f + __expf(-o2))) + fv;
      dtile[afrag_idx(m, col & 31)] = (_Float16)g;
    }
  }
}

// ---- output head: relu(dfeat @ out_W1[p] + b1) . out_W2[p] + b2 + latest ----
__global__ void __launch_bounds__(256) out_kernel(
    const _Float16* __restrict__ dfeatF, const _Float16* __restrict__ ow1F,
    const float* __restrict__ out_b1, const float* __restrict__ out_W2,
    const float* __restrict__ out_b2, const float* __restrict__ x,
    float* __restrict__ out) {
  int nt = blockIdx.x, p = blockIdx.y, b = blockIdx.z;
  int n0 = nt * 16;
  int tid = threadIdx.x, lane = tid & 31, w = tid >> 5;  // 8 waves
  __shared__ float red[16][OUTK];
  __shared__ float sred[16][8];
  const _Float16* Abase = dfeatF + (size_t)(b * NTILE + nt) * 24 * 512;
  const _Float16* Bbase = ow1F + (size_t)p * 24 * 8 * 512;
  v8f acc = {};
  for (int kc = 0; kc < 24; ++kc) {
    v16h a = frag_ld(Abase + (size_t)kc * 512);
    v16h bb = frag_ld(Bbase + (size_t)(kc * 8 + w) * 512);
    acc = wmma16(a, bb, acc);
  }
  int hi = lane >> 4, nc = lane & 15;
  int col = w * 16 + nc;
  float b1v = out_b1[(size_t)p * OUTK + col];
  float w2v = out_W2[(size_t)p * OUTK + col];
#pragma unroll
  for (int r = 0; r < 8; ++r) {
    int m = r + 8 * hi;
    red[m][col] = fmaxf(acc[r] + b1v, 0.f) * w2v;
  }
  __syncthreads();
  if (tid < 128) {
    int m = tid >> 3, seg = tid & 7;
    float s = 0.f;
#pragma unroll
    for (int q = 0; q < 16; ++q) s += red[m][seg * 16 + q];
    sred[m][seg] = s;
  }
  __syncthreads();
  if (tid < 16) {
    int m = tid, n = n0 + m;
    if (n < N_) {
      float s = 0.f;
#pragma unroll
      for (int q = 0; q < 8; ++q) s += sred[m][q];
      float latest = x[(size_t)b * N_ * T_ + (size_t)n * T_ + (T_ - 1)];
      out[((size_t)b * P_ + p) * N_ + n] = s + out_b2[p] + latest;
    }
  }
}

// ---------------- host launch ----------------
extern "C" void kernel_launch(void* const* d_in, const int* in_sizes, int n_in,
                              void* d_out, int out_size, void* d_ws, size_t ws_size,
                              hipStream_t stream) {
  const float* x        = (const float*)d_in[0];
  const float* t_hour   = (const float*)d_in[1];
  const float* t_day    = (const float*)d_in[2];
  const float* W_data   = (const float*)d_in[3];
  const float* b_data   = (const float*)d_in[4];
  const float* W_data1  = (const float*)d_in[5];
  const float* b_data1  = (const float*)d_in[6];
  const float* W_hour   = (const float*)d_in[7];
  const float* b_hour   = (const float*)d_in[8];
  const float* W_day    = (const float*)d_in[9];
  const float* b_day    = (const float*)d_in[10];
  const float* W_tcn    = (const float*)d_in[11];
  const float* b_tcn    = (const float*)d_in[12];
  const float* aml_W1   = (const float*)d_in[13];
  const float* aml_b1   = (const float*)d_in[14];
  const float* aml_W2   = (const float*)d_in[15];
  const float* aml_b2   = (const float*)d_in[16];
  const float* last_W1  = (const float*)d_in[17];
  const float* last_b1  = (const float*)d_in[18];
  const float* last_W2  = (const float*)d_in[19];
  const float* last_b2  = (const float*)d_in[20];
  const float* gc_K     = (const float*)d_in[21];
  const float* gc_b     = (const float*)d_in[22];
  const float* gc_K_last= (const float*)d_in[23];
  const float* gc_b_last= (const float*)d_in[24];
  const float* out_W1   = (const float*)d_in[25];
  const float* out_b1   = (const float*)d_in[26];
  const float* out_W2   = (const float*)d_in[27];
  const float* out_b2   = (const float*)d_in[28];
  float* out = (float*)d_out;

  char* ws = (char*)d_ws;
  size_t off = 0;
  auto carve = [&](size_t bytes) -> void* {
    void* p = ws + off;
    off = (off + bytes + 255) & ~(size_t)255;
    return p;
  };
  const size_t embF_n  = (size_t)B_ * T_ * NTILE * 2 * 512;   // A-frag tiles
  const size_t dT16F_n = (size_t)B_ * T_ * NJ * 4 * 512;      // B-frag tiles
  const size_t dfF_n   = (size_t)B_ * NTILE * 24 * 512;       // A-frag tiles
  _Float16* embF   = (_Float16*)carve(embF_n * 2);
  float*    dT32   = (float*)   carve((size_t)B_ * T_ * N_ * HID * 4);
  _Float16* dT16F  = (_Float16*)carve(dT16F_n * 2);
  _Float16* dfeatF = (_Float16*)carve(dfF_n * 2);
  _Float16* wd1F   = (_Float16*)carve((size_t)2 * 4 * 512 * 2);
  _Float16* wt2F   = (_Float16*)carve((size_t)6 * 4 * 512 * 2);
  _Float16* gckF   = (_Float16*)carve((size_t)TM1 * 16 * 512 * 2);
  _Float16* gcklF  = (_Float16*)carve((size_t)16 * 512 * 2);
  _Float16* ow1F   = (_Float16*)carve((size_t)P_ * 24 * 8 * 512 * 2);
  float*    hd     = (float*)   carve((size_t)B_ * T_ * HID * 4);
  float*    h1     = (float*)   carve((size_t)B_ * TM1 * DD * 4);
  float*    h1l    = (float*)   carve((size_t)B_ * DD * 4);

  // 0) zero the B-fragment feature buffer (K-dim pads j=716..735 must be 0)
  hipMemsetAsync(dT16F, 0, dT16F_n * 2, stream);

  // 1) weight conversion into fragment order
  cvt_bfrag_kernel<<<(64 * 64 + 255) / 256, 256, 0, stream>>>(W_data1, wd1F, 64, 64, 64 * 64);
  cvt_wtcn_frag_kernel<<<(64 * 64 * 3 + 255) / 256, 256, 0, stream>>>(W_tcn, wt2F);
  cvt_bfrag_kernel<<<(TM1 * 64 * 128 + 255) / 256, 256, 0, stream>>>(gc_K, gckF, 64, 128, TM1 * 64 * 128);
  cvt_bfrag_kernel<<<(64 * 128 + 255) / 256, 256, 0, stream>>>(gc_K_last, gcklF, 64, 128, 64 * 128);
  cvt_bfrag_kernel<<<(P_ * TH * 128 + 255) / 256, 256, 0, stream>>>(out_W1, ow1F, TH, 128, P_ * TH * 128);

  // 2) small dense reductions
  hd_kernel<<<B_ * T_, HID, 0, stream>>>(t_hour, t_day, W_hour, b_hour, W_day, b_day, hd);
  h1_kernel<<<dim3(B_, T_), 128, 0, stream>>>(x, aml_W1, aml_b1, last_W1, last_b1, h1, h1l);

  // 3) embedding MLP (WMMA; writes A-fragment tiles)
  embed_kernel<<<dim3(NTILE, B_ * T_), 128, 0, stream>>>(x, wd1F, W_data, b_data,
                                                         b_data1, hd, embF);
  // 4) TCN as K=192 matmul (WMMA; zero LDS, all operands fragment-ordered)
  tcn_kernel<<<dim3(NTILE, B_ * T_), 128, 0, stream>>>(embF, wt2F, b_tcn, dT32, dT16F);

  // 5) fused adjacency -> graph conv -> GLU gate (WMMA), b innermost for L2 reuse
  gate_kernel<<<dim3(NTILE, B_, T_), 128, 0, stream>>>(
      aml_W2, aml_b2, last_W2, last_b2, h1, h1l, gckF, gc_b, gcklF, gc_b_last,
      dT32, dT16F, dfeatF);

  // 6) output head (WMMA; zero staging LDS, fused 128->1 reduction + residual)
  out_kernel<<<dim3(NTILE, P_, B_), 256, 0, stream>>>(dfeatF, ow1F, out_b1, out_W2,
                                                      out_b2, x, out);
}
